// WindowAttention_13846974562647
// MI455X (gfx1250) — matrix operand order repaired and checked by reading, not compiled, not run on hardware
//
#include <hip/hip_runtime.h>
#include <hip/hip_bf16.h>
#include <stdint.h>

typedef __attribute__((ext_vector_type(16))) _Float16 v16h;
typedef __attribute__((ext_vector_type(8)))  float    v8f;

#define N_TOK 64
#define C_CH  192
#define H_HEADS 6
#define HD 32
#define NW 64

// LDS carve offsets (bytes). pm at 0 so epilogue ds_load offsets fit the 16-bit immediate.
#define PM_OFF 0                         // 16384 B : [31:16]=mask f16, [15:0]=rpi*6
#define Q_OFF  16384                     // 24576 B : Q f16, pre-scaled
#define K_OFF  40960                     // 24576 B : K f16
#define VT_OFF 65536                     // 24576 B : V^T f16
#define A_OFF  90112                     // 49152 B : softmaxed probs f16
#define X_OFF  139264                    // 24576 B : attn out f16
#define SMEM_BYTES 163840                // exactly 160 KB -> 2 WGs per 320 KB WGP

// ---- fragment loaders (CDNA5 WMMA 16x16x32 f16, contiguous k = 16*half + s in both A and B) ----

// A matrix 16x32 (MxK) from row-major f16 LDS.
// lanes 0-15: M=lane, K = k0..k0+15 ; lanes 16-31: K = k0+16..k0+31.
__device__ inline v16h load_frag_a(const _Float16* base, int ld, int row0, int k0, int lane) {
    int half = lane >> 4, m = lane & 15;
    const _Float16* row = base + (row0 + m) * ld + k0 + 16 * half;
    v16h f;
#pragma unroll
    for (int v = 0; v < 4; ++v) { f[2*v] = row[2*v]; f[2*v+1] = row[2*v+1]; }
    const _Float16* row2 = row + 8;
#pragma unroll
    for (int v = 0; v < 4; ++v) { f[8+2*v] = row2[2*v]; f[8+2*v+1] = row2[2*v+1]; }
    return f;
}

// B matrix 32x16 (KxN), stored transposed: T[n][k] row-major (T row = one B column).
// lanes 0-15 hold K=k0..k0+15 for N=lane; lanes 16-31 hold K=k0+16..k0+31.
// All offsets are multiples of 16 halfs (32B) -> one aligned v16h LDS read.
__device__ inline v16h load_frag_b(const _Float16* baseT, int ldT, int n0, int k0, int lane) {
    int half = lane >> 4, n = lane & 15;
    const _Float16* row = baseT + (n0 + n) * ldT + k0 + 16 * half;
    return *(const v16h*)row;
}

// B fragment of proj_w^T from global f32 proj_w (row-major [out][in] == B column-major).
// Offsets are multiples of 16 floats -> four aligned b128 loads.
__device__ inline v16h load_frag_b_w(const float* __restrict__ W, int n0, int k0, int lane) {
    int half = lane >> 4, n = lane & 15;
    const float4* row = (const float4*)(W + (n0 + n) * C_CH + k0 + 16 * half);
    v16h f;
#pragma unroll
    for (int q = 0; q < 4; ++q) {
        float4 x = row[q];
        f[4*q+0] = (_Float16)x.x; f[4*q+1] = (_Float16)x.y;
        f[4*q+2] = (_Float16)x.z; f[4*q+3] = (_Float16)x.w;
    }
    return f;
}

__device__ inline uint32_t pack_pm(float m, int idx) {
    unsigned short mb = __builtin_bit_cast(unsigned short, (_Float16)m);
    return ((uint32_t)mb << 16) | (uint32_t)(idx * H_HEADS);   // idx*6 < 1350, fits 16b
}

__global__ __launch_bounds__(192) void win_attn_fused(
    const float* __restrict__ qkv, const int* __restrict__ rpi,
    const float* __restrict__ mask, const float* __restrict__ rpb,
    const float* __restrict__ projw, const float* __restrict__ projb,
    float* __restrict__ out) {

    __shared__ __align__(16) uint8_t smem[SMEM_BYTES];
    uint32_t*  pm = (uint32_t*)(smem + PM_OFF);
    _Float16*  qp = (_Float16*)(smem + Q_OFF);
    _Float16*  kp = (_Float16*)(smem + K_OFF);
    _Float16*  vtp = (_Float16*)(smem + VT_OFF);
    _Float16*  ap = (_Float16*)(smem + A_OFF);
    _Float16*  xp = (_Float16*)(smem + X_OFF);

    const int b    = blockIdx.x;
    const int tid  = threadIdx.x;
    const int lane = tid & 31;
    const int wave = tid >> 5;
    const float scale = 0.17677669529663687f;        // 32^-0.5
    const int widx = b & (NW - 1);

    // ---------------- Stage 1: stage qkv window + packed mask/index into LDS ----------------
    const float4* src4 = (const float4*)(qkv + (size_t)b * N_TOK * 3 * C_CH);
    for (int idx4 = tid; idx4 < (N_TOK * 3 * C_CH) / 4; idx4 += 192) {
        int idx = idx4 * 4;
        int n = idx / (3 * C_CH);
        int e = idx % (3 * C_CH);
        int s = e / C_CH;
        int r = e % C_CH;
        int hh = r / HD, d = r % HD;              // d is 4-aligned; chunk stays in row
        float4 f = src4[idx4];
        if (s == 0) {
            _Float16* dst = qp + (hh * N_TOK + n) * HD + d;
            dst[0] = (_Float16)(f.x * scale); dst[1] = (_Float16)(f.y * scale);
            dst[2] = (_Float16)(f.z * scale); dst[3] = (_Float16)(f.w * scale);
        } else if (s == 1) {
            _Float16* dst = kp + (hh * N_TOK + n) * HD + d;
            dst[0] = (_Float16)f.x; dst[1] = (_Float16)f.y;
            dst[2] = (_Float16)f.z; dst[3] = (_Float16)f.w;
        } else {
            _Float16* dst = vtp + (hh * HD + d) * N_TOK + n;
            dst[0 * N_TOK] = (_Float16)f.x; dst[1 * N_TOK] = (_Float16)f.y;
            dst[2 * N_TOK] = (_Float16)f.z; dst[3 * N_TOK] = (_Float16)f.w;
        }
    }
    const float4* m4 = (const float4*)(mask + (size_t)widx * N_TOK * N_TOK);
    const int4*   r4 = (const int4*)rpi;
    for (int idx4 = tid; idx4 < (N_TOK * N_TOK) / 4; idx4 += 192) {
        float4 mv = m4[idx4];
        int4   rv = r4[idx4];
        uint32_t* pd = pm + idx4 * 4;
        pd[0] = pack_pm(mv.x, rv.x); pd[1] = pack_pm(mv.y, rv.y);
        pd[2] = pack_pm(mv.z, rv.z); pd[3] = pack_pm(mv.w, rv.w);
    }
    __syncthreads();

    // ---------------- Stage 2: attention, one wave per head ----------------
    const int h = wave;                              // 6 waves == 6 heads
    const _Float16* qh = qp + h * N_TOK * HD;
    const _Float16* kh = kp + h * N_TOK * HD;
    const int half = lane >> 4, nc = lane & 15;

    for (int m = 0; m < 4; ++m) {
        v16h afr = load_frag_a(qh, HD, m * 16, 0, lane);
        v8f c[4];
#pragma unroll
        for (int n = 0; n < 4; ++n) {
            v16h bfr = load_frag_b(kh, HD, n * 16, 0, lane);
            v8f z = {};
            c[n] = __builtin_amdgcn_wmma_f32_16x16x32_f16(
                false, afr, false, bfr, (short)0, z, false, false);
        }
        // bias gather: uniform SGPR base + lane index (scale_offset); mask from packed LDS word
#pragma unroll
        for (int r = 0; r < 8; ++r) {
            int i = m * 16 + r + 8 * half;
#pragma unroll
            for (int n = 0; n < 4; ++n) {
                uint32_t u = pm[i * N_TOK + n * 16 + nc];
                float mv = (float)__builtin_bit_cast(_Float16, (unsigned short)(u >> 16));
                c[n][r] += rpb[(u & 0xffffu) + h] + mv;
            }
        }
        // in-register softmax: each row lives in the 16 lanes of a half, 4 tiles wide
#pragma unroll
        for (int r = 0; r < 8; ++r) {
            float mx = fmaxf(fmaxf(c[0][r], c[1][r]), fmaxf(c[2][r], c[3][r]));
            for (int sft = 8; sft >= 1; sft >>= 1) mx = fmaxf(mx, __shfl_xor(mx, sft, 32));
            float sum = 0.f;
#pragma unroll
            for (int n = 0; n < 4; ++n) { c[n][r] = __expf(c[n][r] - mx); sum += c[n][r]; }
            for (int sft = 8; sft >= 1; sft >>= 1) sum += __shfl_xor(sum, sft, 32);
            float inv = 1.0f / sum;
            int i = m * 16 + r + 8 * half;
#pragma unroll
            for (int n = 0; n < 4; ++n)
                ap[(h * N_TOK + i) * N_TOK + n * 16 + nc] = (_Float16)(c[n][r] * inv);
        }
    }

    // AV: (64x64) @ (64x32), K=64 -> two chained WMMAs per 16x16 tile
    const _Float16* ah  = ap + h * N_TOK * N_TOK;
    const _Float16* vth = vtp + h * HD * N_TOK;
    for (int m = 0; m < 4; ++m) {
        v16h a0 = load_frag_a(ah, N_TOK, m * 16, 0,  lane);
        v16h a1 = load_frag_a(ah, N_TOK, m * 16, 32, lane);
#pragma unroll
        for (int n = 0; n < 2; ++n) {
            v16h b0 = load_frag_b(vth, N_TOK, n * 16, 0,  lane);
            v16h b1 = load_frag_b(vth, N_TOK, n * 16, 32, lane);
            v8f z = {};
            v8f c = __builtin_amdgcn_wmma_f32_16x16x32_f16(
                false, a0, false, b0, (short)0, z, false, false);
            c = __builtin_amdgcn_wmma_f32_16x16x32_f16(
                false, a1, false, b1, (short)0, c, false, false);
#pragma unroll
            for (int r = 0; r < 8; ++r) {
                int i = m * 16 + r + 8 * half;
                xp[i * C_CH + h * HD + n * 16 + nc] = (_Float16)c[r];
            }
        }
    }
    __syncthreads();

    // ---------------- Stage 3: output projection (64x192) @ W^T + bias ----------------
    // 48 tiles of 16x16, K=192 (6 WMMA k-steps); W streamed from L2-resident global.
    float* orow = out + (size_t)b * N_TOK * C_CH;
    for (int t = wave; t < 48; t += 6) {
        int m = t / 12, nn = t % 12;
        v8f acc = {};
#pragma unroll
        for (int kb = 0; kb < 6; ++kb) {
            v16h afr = load_frag_a(xp, C_CH, m * 16, kb * 32, lane);
            v16h bfr = load_frag_b_w(projw, nn * 16, kb * 32, lane);
            acc = __builtin_amdgcn_wmma_f32_16x16x32_f16(
                false, afr, false, bfr, (short)0, acc, false, false);
        }
        int j = nn * 16 + nc;
        float pb = projb[j];
#pragma unroll
        for (int r = 0; r < 8; ++r) {
            int i = m * 16 + r + 8 * half;
            orow[i * C_CH + j] = acc[r] + pb;
        }
    }
}

extern "C" void kernel_launch(void* const* d_in, const int* in_sizes, int n_in,
                              void* d_out, int out_size, void* d_ws, size_t ws_size,
                              hipStream_t stream) {
    const float* qkv   = (const float*)d_in[0];
    const int*   rpi   = (const int*)d_in[1];
    const float* mask  = (const float*)d_in[2];
    const float* rpb   = (const float*)d_in[3];
    const float* projw = (const float*)d_in[4];
    const float* projb = (const float*)d_in[5];
    float* out = (float*)d_out;

    int nWin = in_sizes[0] / (N_TOK * 3 * C_CH);   // 4096
    dim3 grid(nWin), block(192);
    hipLaunchKernelGGL(win_attn_fused, grid, block, 0, stream,
                       qkv, rpi, mask, rpb, projw, projb, out);
}
